// KoopmanOperator_26568667693597
// MI455X (gfx1250) — matrix operand order repaired and checked by reading, not compiled, not run on hardware
//
#include <hip/hip_runtime.h>

typedef __attribute__((ext_vector_type(16))) _Float16 v16h;
typedef __attribute__((ext_vector_type(8)))  float    v8f;
typedef _Float16 half_t;

#define KDIM 64
#define LCC  48           // 2*NC
#define NRR  16
#define DTC  0.01f
#define WAVES 4
#define RPW  16           // rows per wave

#define LOG2E     1.4426950408889634f
#define LOG2E2    2.8853900817779268f     // 2*log2(e)
#define DT_LOG2E  (DTC * LOG2E)
#define INV2PI    0.15915494309189535f
#define DT_INV2PI (DTC * INV2PI)

__device__ __forceinline__ v8f wmma16(v16h a, v16h b, v8f c) {
  // D = A(16x32 f16) * B(32x16 f16) + C(16x16 f32)
  return __builtin_amdgcn_wmma_f32_16x16x32_f16(false, a, false, b, (short)0, c,
                                                false, false);
}

// Lane-xor-1 exchange via ds_swizzle SWAPX1 (group-of-32: and=0x1f, or=0, xor=1).
__device__ __forceinline__ float swap_x1(float v) {
  int i = __builtin_bit_cast(int, v);
  i = __builtin_amdgcn_ds_swizzle(i, 0x041F);
  return __builtin_bit_cast(float, i);
}

// Saturating fast tanh(x + b) with pre-scaled bias b2l = 2*log2e*b:
// tanh(x+b) = 1 - 2/(2^(2*log2e*x + b2l) + 1).  Correct +-1 limits, no NaN path.
__device__ __forceinline__ float fast_tanh2(float x, float b2l) {
  const float e2x = __builtin_amdgcn_exp2f(__builtin_fmaf(LOG2E2, x, b2l));
  return 1.0f - 2.0f * __builtin_amdgcn_rcpf(e2x + 1.0f);
}

// Build one 32x16 (KxN) B tile in the CDNA5 B layout:
// lane L holds column N = n0 + (L&15); halves 0..7 -> K = k0 + 8*(L>>4) + h; halves 8..15 -> +16.
__device__ __forceinline__ v16h load_btile(const float* __restrict__ W, int ldk,
                                           int n0, int k0, int kvalid, int lane) {
  const int n  = n0 + (lane & 15);
  const int kb = k0 + 8 * (lane >> 4);
  v16h b;
#pragma unroll
  for (int h = 0; h < 16; ++h) {
    const int k = kb + (h < 8 ? h : h + 8);
    b[h] = (half_t)((k < kvalid) ? W[n * ldk + k] : 0.0f);
  }
  return b;
}

// A tile (16x32, f16) from an LDS f32 row-major [16][64] buffer.
// lane L: M = L&15; halves 0..7 -> K = k0 + 8*(L>>4) + h; halves 8..15 -> +16.
__device__ __forceinline__ v16h load_atile_f32(const float* ybase, int k0, int lane) {
  const int m  = lane & 15;
  const int kb = k0 + 8 * (lane >> 4);
  const float* r = ybase + m * KDIM;
  v16h a;
#pragma unroll
  for (int h = 0; h < 8; ++h) a[h]     = (half_t)r[kb + h];
#pragma unroll
  for (int h = 0; h < 8; ++h) a[8 + h] = (half_t)r[kb + 16 + h];
  return a;
}

// Same, from an LDS f16 row-major buffer with row stride ld.
__device__ __forceinline__ v16h load_atile_f16(const half_t* hbase, int ld, int k0,
                                               int lane) {
  const int m  = lane & 15;
  const int kb = k0 + 8 * (lane >> 4);
  const half_t* r = hbase + m * ld;
  v16h a;
#pragma unroll
  for (int h = 0; h < 8; ++h) a[h]     = r[kb + h];
#pragma unroll
  for (int h = 0; h < 8; ++h) a[8 + h] = r[kb + 16 + h];
  return a;
}

__global__ __launch_bounds__(WAVES * 32)
void koopman_persistent_kernel(const float* __restrict__ x,
                               const float* __restrict__ Wc1, const float* __restrict__ bc1,
                               const float* __restrict__ Wc2, const float* __restrict__ bc2,
                               const float* __restrict__ Wr1, const float* __restrict__ br1,
                               const float* __restrict__ Wr2, const float* __restrict__ br2,
                               float* __restrict__ out, int T) {
  __shared__ float  ysh [WAVES][2][RPW][KDIM]; // 32 KB: ping-pong fp32 state tiles
  __shared__ half_t h1sh[WAVES][RPW][KDIM];    //  8 KB: tanh hidden (cols 48..63 zero)
  __shared__ half_t hrsh[WAVES][RPW][32];      //  4 KB: tanh hidden r (cols 16..31 zero)

  const int lane = threadIdx.x & 31;
  const int wave = threadIdx.x >> 5;
  const int rowbase = blockIdx.x * (WAVES * RPW) + wave * RPW;

  half_t (*H1)[KDIM] = h1sh[wave];
  half_t (*HR)[32]   = hrsh[wave];

  // ---- init: y0 = x[:,0,:] into buffer 0 (wave-private, no barrier) ----
#pragma unroll
  for (int j = 0; j < 8; ++j) {
    const int idx = (lane + 32 * j) * 4;            // 1024 floats total
    const int r = idx >> 6, c = idx & 63;
    const float4 v = *(const float4*)(x + (size_t)(rowbase + r) * KDIM + c);
    *(float4*)(&ysh[wave][0][r][c]) = v;
  }
  // zero the K-padding regions once (never written again)
  for (int j = lane; j < RPW * 16; j += 32) {
    const int r = j >> 4, c = j & 15;
    H1[r][LCC + c] = (half_t)0.0f;                  // cols 48..63
    HR[r][NRR + c] = (half_t)0.0f;                  // cols 16..31
  }

  // ---- weights -> registers, in WMMA B layout (step-invariant) ----
  v16h Bc1[2][3], Bc2[2][3], Br1[2], Br2;
#pragma unroll
  for (int kt = 0; kt < 2; ++kt)
#pragma unroll
    for (int nt = 0; nt < 3; ++nt) {
      Bc1[kt][nt] = load_btile(Wc1, KDIM, nt * 16, kt * 32, KDIM, lane); // 64x48
      Bc2[kt][nt] = load_btile(Wc2, LCC,  nt * 16, kt * 32, LCC,  lane); // 48x48 (K pad->64)
    }
#pragma unroll
  for (int kt = 0; kt < 2; ++kt)
    Br1[kt] = load_btile(Wr1, KDIM, 0, kt * 32, KDIM, lane);             // 64x16
  Br2 = load_btile(Wr2, NRR, 0, 0, NRR, lane);                           // 16x16 (K pad->32)

  const int colQ = lane & 15;      // D-layout column within a 16-tile
  const int rhi  = 8 * (lane >> 4);
  const bool odd = (lane & 1) != 0;

  // per-lane pre-scaled biases for the D layout (col = lane&15 + 16*nt)
  float bc1v2[3], bc2v[3];
#pragma unroll
  for (int nt = 0; nt < 3; ++nt) {
    bc1v2[nt] = LOG2E2 * bc1[nt * 16 + colQ];   // feeds exp2 in fast_tanh2
    bc2v[nt]  = bc2[nt * 16 + colQ];            // raw bias, scaled post-swizzle
  }
  const float br1v2  = LOG2E2  * br1[colQ];
  const float dtbr2l = DT_LOG2E * br2[colQ];    // feeds exp2 for new_r

  // async-store geometry: chunk = j*32+lane; row = chunk>>4; 16B per chunk
  unsigned rowoff[8];   // byte offset of chunk start in `out` at t=0
#pragma unroll
  for (int j = 0; j < 8; ++j) {
    const int chunk = j * 32 + lane;
    const int row = chunk >> 4;
    const int c4  = (chunk & 15) * 4;               // starting float column
    rowoff[j] = (unsigned)(((size_t)(rowbase + row) * T) * KDIM + c4) * 4u;
  }
  const unsigned long long outbase = (unsigned long long)(uintptr_t)out;

  for (int t = 0; t < T; ++t) {
    float (*Ycur)[KDIM] = ysh[wave][t & 1];
    float (*Ynxt)[KDIM] = ysh[wave][(t + 1) & 1];

    // ---- A tiles of current state ----
    const v16h a0 = load_atile_f32(&Ycur[0][0], 0,  lane);
    const v16h a1 = load_atile_f32(&Ycur[0][0], 32, lane);

    // ---- GEMM r1: tanh(y @ Wr1^T + br1) ----
    v8f accr = {};
    accr = wmma16(a0, Br1[0], accr);
    accr = wmma16(a1, Br1[1], accr);
#pragma unroll
    for (int r = 0; r < 8; ++r)
      HR[r + rhi][colQ] = (half_t)fast_tanh2(accr[r], br1v2);

    // ---- GEMM c1: tanh(y @ Wc1^T + bc1) ----
#pragma unroll
    for (int nt = 0; nt < 3; ++nt) {
      v8f acc = {};
      acc = wmma16(a0, Bc1[0][nt], acc);
      acc = wmma16(a1, Bc1[1][nt], acc);
      const int col = nt * 16 + colQ;
#pragma unroll
      for (int r = 0; r < 8; ++r)
        H1[r + rhi][col] = (half_t)fast_tanh2(acc[r], bc1v2[nt]);
    }
    asm volatile("" ::: "memory");  // keep DS stores before reloads (HW DS is in-order per wave)

    // ---- second-stage A tiles ----
    const v16h ha0 = load_atile_f16(&H1[0][0], KDIM, 0,  lane);
    const v16h ha1 = load_atile_f16(&H1[0][0], KDIM, 32, lane);
    const v16h hra = load_atile_f16(&HR[0][0], 32,   0,  lane);

    // ---- GEMM r2 + GEMM c2 ----
    v8f accre = {};
    accre = wmma16(hra, Br2, accre);

    float newc[3][8];
#pragma unroll
    for (int nt = 0; nt < 3; ++nt) {
      v8f acc = {};
      acc = wmma16(ha0, Bc2[0][nt], acc);
      acc = wmma16(ha1, Bc2[1][nt], acc);
      const int col = nt * 16 + colQ;          // parity(col) == parity(lane)
#pragma unroll
      for (int r = 0; r < 8; ++r) {
        const float v = acc[r] + bc2v[nt];     // raw co value (mu or omega)
        const float w = swap_x1(v);            // partner column (ds_swizzle SWAPX1)
        const float mu = odd ? w : v;
        const float om = odd ? v : w;
        const float e   = __builtin_amdgcn_exp2f(mu * DT_LOG2E);  // e^(dt*mu)
        const float ang = om * DT_INV2PI;                         // revolutions
        const float cc  = __builtin_amdgcn_cosf(ang) * e;
        const float ss  = __builtin_amdgcn_sinf(ang) * e;
        const int row = r + rhi;
        const float y0 = Ycur[row][col & ~1];
        const float y1 = Ycur[row][col |  1];
        newc[nt][r] = odd ? (ss * y0 + cc * y1) : (cc * y0 - ss * y1);
      }
    }
    float newr[8];
    {
      const int col = LCC + colQ;
#pragma unroll
      for (int r = 0; r < 8; ++r)
        newr[r] = __builtin_amdgcn_exp2f(__builtin_fmaf(DT_LOG2E, accre[r], dtbr2l)) *
                  Ycur[r + rhi][col];
    }

    // Ynxt was async-stored 2 steps ago; in-order async completion means
    // asynccnt<=8 guarantees those 8 ops drained before we overwrite it.
    asm volatile("s_wait_asynccnt 8" ::: "memory");

    // ---- commit next state to the ping-pong buffer ----
#pragma unroll
    for (int nt = 0; nt < 3; ++nt) {
      const int col = nt * 16 + colQ;
#pragma unroll
      for (int r = 0; r < 8; ++r)
        Ynxt[r + rhi][col] = newc[nt][r];
    }
    {
      const int col = LCC + colQ;
#pragma unroll
      for (int r = 0; r < 8; ++r)
        Ynxt[r + rhi][col] = newr[r];
    }

    // make DS-pipe commit visible to the async DMA engine, then fire the
    // LDS->HBM async stores for this step's output row block (B,T,64)
    asm volatile("s_wait_dscnt 0" ::: "memory");
#pragma unroll
    for (int j = 0; j < 8; ++j) {
      const int chunk = j * 32 + lane;
      const int row = chunk >> 4;
      const int c4  = (chunk & 15) * 4;
      const unsigned ldsa = (unsigned)(uintptr_t)(&Ynxt[row][c4]);
      const unsigned voff = rowoff[j] + (unsigned)t * (KDIM * 4);
      asm volatile("global_store_async_from_lds_b128 %0, %1, %2"
                   :: "v"(voff), "v"(ldsa), "s"(outbase) : "memory");
    }
  }
  // s_endpgm performs an implicit wait-idle: all async stores drain before exit
}

extern "C" void kernel_launch(void* const* d_in, const int* in_sizes, int n_in,
                              void* d_out, int out_size, void* d_ws, size_t ws_size,
                              hipStream_t stream) {
  const float* x   = (const float*)d_in[0];
  const float* Wc1 = (const float*)d_in[1];
  const float* bc1 = (const float*)d_in[2];
  const float* Wc2 = (const float*)d_in[3];
  const float* bc2 = (const float*)d_in[4];
  const float* Wr1 = (const float*)d_in[5];
  const float* br1 = (const float*)d_in[6];
  const float* Wr2 = (const float*)d_in[7];
  const float* br2 = (const float*)d_in[8];
  float* out = (float*)d_out;

  const int B = in_sizes[0] / KDIM;            // x is (B,1,64)
  const int T = out_size / in_sizes[0];        // out is (B,T,64)
  const int rows_per_block = WAVES * RPW;      // 64
  const int grid = B / rows_per_block;         // 4096/64 = 64 blocks of 4 waves

  koopman_persistent_kernel<<<grid, WAVES * 32, 0, stream>>>(
      x, Wc1, bc1, Wc2, bc2, Wr1, br1, Wr2, br2, out, T);
}
